// ComplexAttention_84602265796827
// MI455X (gfx1250) — matrix operand order repaired and checked by reading, not compiled, not run on hardware
//
#include <hip/hip_runtime.h>

// ---------------------------------------------------------------------------
// ComplexAttention for MI455X (gfx1250, wave32, WMMA bf16 + TDM async tiles)
// Pipeline: 6x proj GEMM -> scores (dual-accumulate) -> softmax -> 2x PV GEMM
//           -> complex layernorm.
// GEMM core: v_wmma_f32_16x16x32_bf16.
// bf16 tile staging: TENSOR_LOAD_TO_LDS (TDM, TENSORcnt), double-buffered so
// the DMA overlaps WMMA.  f32->bf16 staging: register-pipelined b128 loads.
// ---------------------------------------------------------------------------

typedef __bf16 bf16;
typedef __attribute__((ext_vector_type(16))) __bf16 v16bf;
typedef __attribute__((ext_vector_type(8)))  float  v8f;
typedef __attribute__((ext_vector_type(4)))  unsigned int u32x4;
typedef __attribute__((ext_vector_type(8)))  int i32x8;
typedef __attribute__((ext_vector_type(4)))  int i32x4;

#define DEVINL __device__ __forceinline__

constexpr int Bb = 16, Ss = 1024, Dd = 1024;
constexpr int BM = 128, BN = 64, BK = 32;   // block tile
constexpr int NTHREADS = 256;               // 8 waves (wave32)

// ---- Tensor Data Mover: 2-D tile global->LDS (ISA 8.3/8.4 D# layout) -------
DEVINL void tdm_load_tile_bf16(unsigned int lds_off, const void* gptr,
                               unsigned int tile_k, unsigned int tile_rows,
                               unsigned int row_stride) {
  unsigned long long ga = (unsigned long long)(uintptr_t)gptr;
  u32x4 g0;
  g0[0] = 1u;                                          // count=1 (user D#)
  g0[1] = lds_off;                                     // lds_addr (bytes)
  g0[2] = (unsigned int)(ga & 0xFFFFFFFFu);            // global_addr[31:0]
  g0[3] = (unsigned int)((ga >> 32) & 0x01FFFFFFu)     // global_addr[56:32]
        | (2u << 30);                                  // type=2 ("image")
  i32x8 g1;
  g1[0] = (int)(1u << 16);                             // data_size=1 (2 bytes)
  g1[1] = (int)((tile_k & 0xFFFFu) << 16);             // tensor_dim0[15:0]
  g1[2] = (int)(((tile_k >> 16) & 0xFFFFu)             // tensor_dim0[31:16]
        | ((tile_rows & 0xFFFFu) << 16));              // tensor_dim1[15:0]
  g1[3] = (int)(((tile_rows >> 16) & 0xFFFFu)          // tensor_dim1[31:16]
        | ((tile_k & 0xFFFFu) << 16));                 // tile_dim0
  g1[4] = (int)(tile_rows & 0xFFFFu);                  // tile_dim1 (tile_dim2=0)
  g1[5] = (int)row_stride;                             // tensor_dim0_stride[31:0]
  g1[6] = 0;
  g1[7] = 0;
  i32x4 z4 = {0, 0, 0, 0};
#if defined(__clang_major__) && (__clang_major__ >= 23)
  i32x8 z8 = {0, 0, 0, 0, 0, 0, 0, 0};
  __builtin_amdgcn_tensor_load_to_lds(g0, g1, z4, z4, z8, 0);
#else
  __builtin_amdgcn_tensor_load_to_lds(g0, g1, z4, z4, 0);
#endif
}
DEVINL unsigned int lds_off_of(const void* p) {        // LDS aperture: addr[31:0]
  return (unsigned int)(uintptr_t)p;
}

// ---- WMMA fragment loaders (CDNA5 ISA 7.12.2 layouts) ----------------------
DEVINL v16bf frag_a(const bf16* lA, int m0, int lane) {
  const int r = lane & 15, h = lane >> 4;
  const bf16* base = lA + (m0 + r) * BK;
  union { uint4 q[2]; v16bf v; } u;
  u.q[0] = *(const uint4*)(base + 8 * h);
  u.q[1] = *(const uint4*)(base + 16 + 8 * h);
  return u.v;
}
DEVINL v16bf frag_b(const bf16* lB, int n0, int lane) {
  const int n = lane & 15, h = lane >> 4;
  const bf16* base = lB + (n0 + n) * BK;
  union { uint4 q[2]; v16bf v; } u;
  u.q[0] = *(const uint4*)(base + 8 * h);
  u.q[1] = *(const uint4*)(base + 16 + 8 * h);
  return u.v;
}
DEVINL v8f wmma_bf16(v16bf a, v16bf b, v8f c) {
  return __builtin_amdgcn_wmma_f32_16x16x32_bf16(false, a, false, b,
                                                 (short)0, c, false, false);
}
DEVINL void cvt_store4(bf16* dst, float4 f) {
  union { bf16 h[4]; uint2 u; } p;
  p.h[0] = (bf16)f.x; p.h[1] = (bf16)f.y; p.h[2] = (bf16)f.z; p.h[3] = (bf16)f.w;
  *(uint2*)dst = p.u;
}
// transpose stage: lds[n][k] = g[(kb+k)*ld + n0+n], 32x64 tile, batched
DEVINL void stage_tr(const bf16* g, bf16* lds, int kb, int n0, int tid, int ld) {
  unsigned short t8[8];
  #pragma unroll
  for (int i = 0; i < 8; ++i) {
    int lin = tid + NTHREADS * i;
    t8[i] = ((const unsigned short*)g)[(size_t)(kb + (lin >> 6)) * ld + n0 + (lin & 63)];
  }
  #pragma unroll
  for (int i = 0; i < 8; ++i) {
    int lin = tid + NTHREADS * i;
    ((unsigned short*)lds)[(lin & 63) * BK + (lin >> 6)] = t8[i];
  }
}

// ---------------------------------------------------------------------------
// K1: Y[m,n] = bf16( sum_k X[m,k] * W[n,k] + bias[n] )   (register-pipelined)
// ---------------------------------------------------------------------------
__global__ __launch_bounds__(NTHREADS)
void proj_gemm_kernel(const float* __restrict__ X, const float* __restrict__ W,
                      const float* __restrict__ bias, bf16* __restrict__ Y,
                      int M, int N, int K) {
  __shared__ __align__(16) bf16 lA[BM * BK];
  __shared__ __align__(16) bf16 lB[BN * BK];
  const int tid = threadIdx.x, lane = tid & 31, wave = tid >> 5;
  const int wm = (wave >> 1) * 32, wn = (wave & 1) * 32;
  const int m0 = blockIdx.y * BM, n0 = blockIdx.x * BN;
  v8f acc[2][2] = {};

  float4 fa[4], fb[2];
  #pragma unroll
  for (int i = 0; i < 4; ++i) {
    int lin = tid + NTHREADS * i;
    fa[i] = *(const float4*)(X + (size_t)(m0 + (lin >> 3)) * K + (lin & 7) * 4);
  }
  #pragma unroll
  for (int i = 0; i < 2; ++i) {
    int lin = tid + NTHREADS * i;
    fb[i] = *(const float4*)(W + (size_t)(n0 + (lin >> 3)) * K + (lin & 7) * 4);
  }

  for (int kb = 0; kb < K; kb += BK) {
    #pragma unroll
    for (int i = 0; i < 4; ++i) {
      int lin = tid + NTHREADS * i;
      cvt_store4(lA + (lin >> 3) * BK + (lin & 7) * 4, fa[i]);
    }
    #pragma unroll
    for (int i = 0; i < 2; ++i) {
      int lin = tid + NTHREADS * i;
      cvt_store4(lB + (lin >> 3) * BK + (lin & 7) * 4, fb[i]);
    }
    __syncthreads();

    // issue next tile's global loads now: they fly during ds_load + WMMA below
    if (kb + BK < K) {
      #pragma unroll
      for (int i = 0; i < 4; ++i) {
        int lin = tid + NTHREADS * i;
        fa[i] = *(const float4*)(X + (size_t)(m0 + (lin >> 3)) * K + kb + BK + (lin & 7) * 4);
      }
      #pragma unroll
      for (int i = 0; i < 2; ++i) {
        int lin = tid + NTHREADS * i;
        fb[i] = *(const float4*)(W + (size_t)(n0 + (lin >> 3)) * K + kb + BK + (lin & 7) * 4);
      }
      if (kb + 2 * BK < K)   // L2 warmup two steps ahead
        __builtin_prefetch(X + (size_t)(m0 + (tid >> 1)) * K + kb + 2 * BK + (tid & 1) * 16, 0, 3);
    }

    v16bf a0 = frag_a(lA, wm, lane), a1 = frag_a(lA, wm + 16, lane);
    v16bf b0 = frag_b(lB, wn, lane), b1 = frag_b(lB, wn + 16, lane);
    acc[0][0] = wmma_bf16(a0, b0, acc[0][0]);
    acc[0][1] = wmma_bf16(a0, b1, acc[0][1]);
    acc[1][0] = wmma_bf16(a1, b0, acc[1][0]);
    acc[1][1] = wmma_bf16(a1, b1, acc[1][1]);
    __syncthreads();
  }
  const int cr = lane & 15, ch = lane >> 4;
  #pragma unroll
  for (int tm = 0; tm < 2; ++tm)
    #pragma unroll
    for (int tn = 0; tn < 2; ++tn) {
      int col = n0 + wn + tn * 16 + cr;
      float bv = bias[col];
      #pragma unroll
      for (int i = 0; i < 8; ++i) {
        int row = m0 + wm + tm * 16 + ch * 8 + i;
        Y[(size_t)row * N + col] = (bf16)(acc[tm][tn][i] + bv);
      }
    }
}

// ---------------------------------------------------------------------------
// K2: Sc[b,s,e] = ( sum_d Qr[s,d]*Kr[d,e] + sum_d Qi[s,d]*Ki[e,d] ) / 8
// Double-buffered: TDM DMAs the next copy tiles while WMMA runs on current.
// ---------------------------------------------------------------------------
__global__ __launch_bounds__(NTHREADS)
void scores_kernel(const bf16* __restrict__ Qr, const bf16* __restrict__ Qi,
                   const bf16* __restrict__ Kr, const bf16* __restrict__ Ki,
                   float* __restrict__ Sc) {
  __shared__ __align__(16) bf16 lAr[2][BM * BK], lAi[2][BM * BK];
  __shared__ __align__(16) bf16 lBr[2][BN * BK], lBi[2][BN * BK];
  const int tid = threadIdx.x, lane = tid & 31, wave = tid >> 5;
  const int wm = (wave >> 1) * 32, wn = (wave & 1) * 32;
  const int m0 = blockIdx.y * BM, n0 = blockIdx.x * BN;
  const size_t boff = (size_t)blockIdx.z * Ss * Dd;
  const bf16* qr = Qr + boff; const bf16* qi = Qi + boff;
  const bf16* kr = Kr + boff; const bf16* ki = Ki + boff;
  float* sc = Sc + (size_t)blockIdx.z * Ss * Ss;
  v8f acc[2][2] = {};

  // prologue: stage buffer 0
  if (wave == 0) {
    tdm_load_tile_bf16(lds_off_of(lAr[0]), qr + (size_t)m0 * Dd, BK, BM, Dd);
    tdm_load_tile_bf16(lds_off_of(lAi[0]), qi + (size_t)m0 * Dd, BK, BM, Dd);
    tdm_load_tile_bf16(lds_off_of(lBi[0]), ki + (size_t)n0 * Dd, BK, BN, Dd);
  }
  stage_tr(kr, lBr[0], 0, n0, tid, Dd);
  if (wave == 0) __builtin_amdgcn_s_wait_tensorcnt(0);
  __syncthreads();

  for (int kb = 0; kb < Dd; kb += BK) {
    const int cur = (kb >> 5) & 1, nxt = cur ^ 1;
    if (kb + BK < Dd) {                           // stage next buffer (async)
      if (wave == 0) {
        tdm_load_tile_bf16(lds_off_of(lAr[nxt]), qr + (size_t)m0 * Dd + kb + BK, BK, BM, Dd);
        tdm_load_tile_bf16(lds_off_of(lAi[nxt]), qi + (size_t)m0 * Dd + kb + BK, BK, BM, Dd);
        tdm_load_tile_bf16(lds_off_of(lBi[nxt]), ki + (size_t)n0 * Dd + kb + BK, BK, BN, Dd);
      }
      stage_tr(kr, lBr[nxt], kb + BK, n0, tid, Dd);
    }
    // compute on current buffer (overlaps the DMA above)
    v16bf ar0 = frag_a(lAr[cur], wm, lane), ar1 = frag_a(lAr[cur], wm + 16, lane);
    v16bf ai0 = frag_a(lAi[cur], wm, lane), ai1 = frag_a(lAi[cur], wm + 16, lane);
    v16bf br0 = frag_b(lBr[cur], wn, lane), br1 = frag_b(lBr[cur], wn + 16, lane);
    v16bf bi0 = frag_b(lBi[cur], wn, lane), bi1 = frag_b(lBi[cur], wn + 16, lane);
    acc[0][0] = wmma_bf16(ar0, br0, acc[0][0]);
    acc[0][0] = wmma_bf16(ai0, bi0, acc[0][0]);
    acc[0][1] = wmma_bf16(ar0, br1, acc[0][1]);
    acc[0][1] = wmma_bf16(ai0, bi1, acc[0][1]);
    acc[1][0] = wmma_bf16(ar1, br0, acc[1][0]);
    acc[1][0] = wmma_bf16(ai1, bi0, acc[1][0]);
    acc[1][1] = wmma_bf16(ar1, br1, acc[1][1]);
    acc[1][1] = wmma_bf16(ai1, bi1, acc[1][1]);
    if (wave == 0) __builtin_amdgcn_s_wait_tensorcnt(0);
    __syncthreads();
  }
  const int cr = lane & 15, ch = lane >> 4;
  #pragma unroll
  for (int tm = 0; tm < 2; ++tm)
    #pragma unroll
    for (int tn = 0; tn < 2; ++tn)
      #pragma unroll
      for (int i = 0; i < 8; ++i) {
        int row = m0 + wm + tm * 16 + ch * 8 + i;
        int col = n0 + wn + tn * 16 + cr;
        sc[(size_t)row * Ss + col] = acc[tm][tn][i] * 0.125f;
      }
}

// ---------------------------------------------------------------------------
// K3: row softmax over Sc (fp32) -> probs P (bf16). One block per row.
// ---------------------------------------------------------------------------
__global__ __launch_bounds__(256)
void softmax_kernel(const float* __restrict__ Sc, bf16* __restrict__ P) {
  const size_t row = blockIdx.x;
  const float* src = Sc + row * Ss;
  bf16* dst = P + row * Ss;
  const int t = threadIdx.x;
  float4 f = *(const float4*)(src + t * 4);

  float m = fmaxf(fmaxf(f.x, f.y), fmaxf(f.z, f.w));
  #pragma unroll
  for (int off = 16; off > 0; off >>= 1) m = fmaxf(m, __shfl_xor(m, off));
  __shared__ float sm[8];
  if ((t & 31) == 0) sm[t >> 5] = m;
  __syncthreads();
  float bm = sm[0];
  #pragma unroll
  for (int i = 1; i < 8; ++i) bm = fmaxf(bm, sm[i]);

  float e0 = __expf(f.x - bm), e1 = __expf(f.y - bm);
  float e2 = __expf(f.z - bm), e3 = __expf(f.w - bm);
  float s = e0 + e1 + e2 + e3;
  #pragma unroll
  for (int off = 16; off > 0; off >>= 1) s += __shfl_xor(s, off);
  __shared__ float ss[8];
  if ((t & 31) == 0) ss[t >> 5] = s;
  __syncthreads();
  float bs = 0.f;
  #pragma unroll
  for (int i = 0; i < 8; ++i) bs += ss[i];
  float inv = 1.0f / bs;

  cvt_store4(dst + t * 4, make_float4(e0 * inv, e1 * inv, e2 * inv, e3 * inv));
}

// ---------------------------------------------------------------------------
// K4: O[b,s,n] = sum_t P[b,s,t] * V[b,t,n]; P tile via double-buffered TDM.
// ---------------------------------------------------------------------------
__global__ __launch_bounds__(NTHREADS)
void pv_kernel(const bf16* __restrict__ P, const bf16* __restrict__ V,
               float* __restrict__ O) {
  __shared__ __align__(16) bf16 lA[2][BM * BK];
  __shared__ __align__(16) bf16 lB[2][BN * BK];
  const int tid = threadIdx.x, lane = tid & 31, wave = tid >> 5;
  const int wm = (wave >> 1) * 32, wn = (wave & 1) * 32;
  const int m0 = blockIdx.y * BM, n0 = blockIdx.x * BN;
  const bf16* p = P + (size_t)blockIdx.z * Ss * Ss;
  const bf16* v = V + (size_t)blockIdx.z * Ss * Dd;
  float* o = O + (size_t)blockIdx.z * Ss * Dd;
  v8f acc[2][2] = {};

  if (wave == 0)
    tdm_load_tile_bf16(lds_off_of(lA[0]), p + (size_t)m0 * Ss, BK, BM, Ss);
  stage_tr(v, lB[0], 0, n0, tid, Dd);
  if (wave == 0) __builtin_amdgcn_s_wait_tensorcnt(0);
  __syncthreads();

  for (int kb = 0; kb < Ss; kb += BK) {
    const int cur = (kb >> 5) & 1, nxt = cur ^ 1;
    if (kb + BK < Ss) {
      if (wave == 0)
        tdm_load_tile_bf16(lds_off_of(lA[nxt]), p + (size_t)m0 * Ss + kb + BK, BK, BM, Ss);
      stage_tr(v, lB[nxt], kb + BK, n0, tid, Dd);
    }
    v16bf a0 = frag_a(lA[cur], wm, lane), a1 = frag_a(lA[cur], wm + 16, lane);
    v16bf b0 = frag_b(lB[cur], wn, lane), b1 = frag_b(lB[cur], wn + 16, lane);
    acc[0][0] = wmma_bf16(a0, b0, acc[0][0]);
    acc[0][1] = wmma_bf16(a0, b1, acc[0][1]);
    acc[1][0] = wmma_bf16(a1, b0, acc[1][0]);
    acc[1][1] = wmma_bf16(a1, b1, acc[1][1]);
    if (wave == 0) __builtin_amdgcn_s_wait_tensorcnt(0);
    __syncthreads();
  }
  const int cr = lane & 15, ch = lane >> 4;
  #pragma unroll
  for (int tm = 0; tm < 2; ++tm)
    #pragma unroll
    for (int tn = 0; tn < 2; ++tn)
      #pragma unroll
      for (int i = 0; i < 8; ++i) {
        int row = m0 + wm + tm * 16 + ch * 8 + i;
        int col = n0 + wn + tn * 16 + cr;
        o[(size_t)row * Dd + col] = acc[tm][tn][i];
      }
}

// ---------------------------------------------------------------------------
// K5: complex layernorm
// ---------------------------------------------------------------------------
__global__ __launch_bounds__(256)
void cln_kernel(const float* __restrict__ Or, const float* __restrict__ Oi,
                const float* __restrict__ a2, const float* __restrict__ b2,
                float* __restrict__ Yr, float* __restrict__ Yi) {
  const size_t row = blockIdx.x;
  const float* ar = Or + row * Dd;
  const float* ai = Oi + row * Dd;
  const int t = threadIdx.x;
  float4 xr = *(const float4*)(ar + t * 4);
  float4 xi = *(const float4*)(ai + t * 4);

  float v5[5];
  v5[0] = xr.x + xr.y + xr.z + xr.w;
  v5[1] = xi.x + xi.y + xi.z + xi.w;
  v5[2] = xr.x*xr.x + xr.y*xr.y + xr.z*xr.z + xr.w*xr.w;
  v5[3] = xi.x*xi.x + xi.y*xi.y + xi.z*xi.z + xi.w*xi.w;
  v5[4] = xr.x*xi.x + xr.y*xi.y + xr.z*xi.z + xr.w*xi.w;

  __shared__ float red[5][8];
  #pragma unroll
  for (int j = 0; j < 5; ++j) {
    float x = v5[j];
    #pragma unroll
    for (int off = 16; off > 0; off >>= 1) x += __shfl_xor(x, off);
    if ((t & 31) == 0) red[j][t >> 5] = x;
  }
  __syncthreads();
  float tot[5];
  #pragma unroll
  for (int j = 0; j < 5; ++j) {
    float s = 0.f;
    #pragma unroll
    for (int i = 0; i < 8; ++i) s += red[j][i];
    tot[j] = s;
  }

  const float invD = 1.0f / (float)Dd;
  float mr = tot[0] * invD, mi = tot[1] * invD;
  float ezr = (tot[2] - tot[3]) * invD;
  float ezi = 2.0f * tot[4] * invD;
  float vr = ezr - (mr * mr - mi * mi);
  float vi = ezi - 2.0f * mr * mi;
  float wr = vr + 1e-6f, wi = vi;
  float rmag = sqrtf(wr * wr + wi * wi);
  float sr = sqrtf(fmaxf(0.f, 0.5f * (rmag + wr)));
  float si = copysignf(sqrtf(fmaxf(0.f, 0.5f * (rmag - wr))), wi);
  float den = sr * sr + si * si;
  float idn = den > 0.f ? 1.0f / den : 0.f;

  const float* pr = (const float*)&xr;
  const float* pi = (const float*)&xi;
  #pragma unroll
  for (int j = 0; j < 4; ++j) {
    int c = t * 4 + j;
    float u = pr[j] - mr, w = pi[j] - mi;
    float dr = (u * sr + w * si) * idn;
    float di = (w * sr - u * si) * idn;
    Yr[row * Dd + c] = a2[c] * dr + b2[c];
    Yi[row * Dd + c] = a2[c] * di;
  }
}

// ---------------------------------------------------------------------------
extern "C" void kernel_launch(void* const* d_in, const int* in_sizes, int n_in,
                              void* d_out, int out_size, void* d_ws, size_t ws_size,
                              hipStream_t stream) {
  const float* q_real = (const float*)d_in[0];
  const float* q_imag = (const float*)d_in[1];
  const float* k_real = (const float*)d_in[2];
  const float* k_imag = (const float*)d_in[3];
  const float* v_real = (const float*)d_in[4];
  const float* v_imag = (const float*)d_in[5];
  const float* Wq_r = (const float*)d_in[6];  const float* bq_r = (const float*)d_in[7];
  const float* Wq_i = (const float*)d_in[8];  const float* bq_i = (const float*)d_in[9];
  const float* Wk_r = (const float*)d_in[10]; const float* bk_r = (const float*)d_in[11];
  const float* Wk_i = (const float*)d_in[12]; const float* bk_i = (const float*)d_in[13];
  const float* Wv_r = (const float*)d_in[14]; const float* bv_r = (const float*)d_in[15];
  const float* Wv_i = (const float*)d_in[16]; const float* bv_i = (const float*)d_in[17];
  const float* a_2  = (const float*)d_in[18]; const float* b_2  = (const float*)d_in[19];

  const size_t nQ = (size_t)Bb * Ss * Dd;
  const size_t nS = (size_t)Bb * Ss * Ss;
  bf16*  Qr = (bf16*)d_ws;
  bf16*  Qi = Qr + nQ;
  bf16*  Kr = Qi + nQ;
  bf16*  Ki = Kr + nQ;
  bf16*  Vr = Ki + nQ;
  bf16*  Vi = Vr + nQ;
  float* Sc = (float*)(Vi + nQ);
  bf16*  P  = (bf16*)(Sc + nS);
  float* Or = (float*)(P + nS);
  float* Oi = Or + nQ;
  float* Yr = (float*)d_out;
  float* Yi = Yr + nQ;

  dim3 blk(NTHREADS);
  dim3 gProj(Dd / BN, (Bb * Ss) / BM);
  proj_gemm_kernel<<<gProj, blk, 0, stream>>>(q_real, Wq_r, bq_r, Qr, Bb * Ss, Dd, Dd);
  proj_gemm_kernel<<<gProj, blk, 0, stream>>>(q_imag, Wq_i, bq_i, Qi, Bb * Ss, Dd, Dd);
  proj_gemm_kernel<<<gProj, blk, 0, stream>>>(k_real, Wk_r, bk_r, Kr, Bb * Ss, Dd, Dd);
  proj_gemm_kernel<<<gProj, blk, 0, stream>>>(k_imag, Wk_i, bk_i, Ki, Bb * Ss, Dd, Dd);
  proj_gemm_kernel<<<gProj, blk, 0, stream>>>(v_real, Wv_r, bv_r, Vr, Bb * Ss, Dd, Dd);
  proj_gemm_kernel<<<gProj, blk, 0, stream>>>(v_imag, Wv_i, bv_i, Vi, Bb * Ss, Dd, Dd);

  dim3 gSc(Ss / BN, Ss / BM, Bb);
  scores_kernel<<<gSc, blk, 0, stream>>>(Qr, Qi, Kr, Ki, Sc);

  softmax_kernel<<<Bb * Ss, 256, 0, stream>>>(Sc, P);

  dim3 gPV(Dd / BN, Ss / BM, Bb);
  pv_kernel<<<gPV, blk, 0, stream>>>(P, Vr, Or);
  pv_kernel<<<gPV, blk, 0, stream>>>(P, Vi, Oi);

  cln_kernel<<<Bb * Ss, 256, 0, stream>>>(Or, Oi, a_2, b_2, Yr, Yi);
}
